// CaSe4SR_49108656062474
// MI455X (gfx1250) — matrix operand order repaired and checked by reading, not compiled, not run on hardware
//
#include <hip/hip_runtime.h>

#define DIM      128
#define D2       256
#define NUM_ITEM 100000
#define N_I      65536
#define N_C      8192
#define E_I      524288
#define E_C      65536
#define T_POS    102400
#define S_SESS   2048
#define V_ALL    NUM_ITEM

typedef __attribute__((ext_vector_type(16))) __bf16 v16bf;
typedef __attribute__((ext_vector_type(8)))  float  v8f;
typedef __attribute__((ext_vector_type(4)))  unsigned int v4u;
typedef __attribute__((ext_vector_type(8)))  unsigned int v8u;

#define WMMA_BF16(a, b, c) \
  __builtin_amdgcn_wmma_f32_16x16x32_bf16(false, (a), false, (b), (short)0, (c), false, false)

// ---- fragment loaders (A and B share one contiguous k order) ----
// A 16x32 bf16: element e -> K = half*16 + e (contiguous halves)
__device__ inline v16bf frag_a_f32(const float* rowk, int half) {
  v16bf r;
  const float* p0 = rowk + half * 16;
  const float* p1 = rowk + half * 16 + 8;
#pragma unroll
  for (int i = 0; i < 8; ++i) { r[i] = (__bf16)p0[i]; r[8 + i] = (__bf16)p1[i]; }
  return r;
}
// B 32x16 bf16: element e -> K = half*16 + e (contiguous halves)
__device__ inline v16bf frag_b_f32(const float* rowk, int half) {
  v16bf r;
  const float* p = rowk + half * 16;
#pragma unroll
  for (int i = 0; i < 16; ++i) r[i] = (__bf16)p[i];
  return r;
}
__device__ inline v16bf frag_a_bf16(const __bf16* rowk, int half) {
  union { v16bf v; uint4 q[2]; } u;
  u.q[0] = *(const uint4*)(rowk + half * 16);
  u.q[1] = *(const uint4*)(rowk + half * 16 + 8);
  return u.v;
}
__device__ inline v16bf frag_b_bf16(const __bf16* rowk, int half) {
  return *(const v16bf*)(rowk + half * 16);
}

// LDS offset of a __shared__ object: low 32 bits of the flat address (LDS aperture).
__device__ inline unsigned lds_off(const void* p) { return (unsigned)(size_t)p; }

// ---- GNN edge scatter: agg[dst] += w * emb_table[node_id[src]] ----
__global__ void gnn_scatter(const float* __restrict__ emb, const int* __restrict__ node_id,
                            const int* __restrict__ esrc, const int* __restrict__ edst,
                            const float* __restrict__ ew, float* __restrict__ agg, int nE) {
  const int e = blockIdx.x * (blockDim.x >> 5) + (threadIdx.x >> 5);
  if (e >= nE) return;
  const int lane = threadIdx.x & 31;
  const float w = ew[e];
  const float4 x = *(const float4*)(emb + (size_t)node_id[esrc[e]] * DIM + lane * 4);
  float* o = agg + (size_t)edst[e] * DIM + lane * 4;
  atomicAdd(o + 0, x.x * w);
  atomicAdd(o + 1, x.y * w);
  atomicAdd(o + 2, x.z * w);
  atomicAdd(o + 3, x.w * w);
}

// ---- out[M,128] = agg[M,128] @ W^T + b   (block: 16 rows x 128 cols, 8 waves) ----
// W (64KB f32) is staged block-wide into LDS with async global->LDS loads (ASYNCcnt path).
__global__ void gnn_linear_wmma(const float* __restrict__ agg, const float* __restrict__ W,
                                const float* __restrict__ bias, float* __restrict__ out) {
  __shared__ float ldsW[DIM * DIM];  // 64 KB
  const unsigned lbase = lds_off(&ldsW[0]);
  // 64KB / (256 threads * 16B) = 16 async b128 copies per thread
#pragma unroll
  for (int it = 0; it < 16; ++it) {
    const unsigned byteoff = (unsigned)(it * 256 + threadIdx.x) * 16u;
    const unsigned la = lbase + byteoff;
    const char* ga = (const char*)W + byteoff;
    asm volatile("global_load_async_to_lds_b128 %0, %1, off" :: "v"(la), "v"(ga) : "memory");
  }
  asm volatile("s_wait_asynccnt 0x0" ::: "memory");
  __syncthreads();

  const int lane = threadIdx.x & 31;
  const int wave = threadIdx.x >> 5;
  const int half = lane >> 4;
  const int lm   = lane & 15;
  const int m0 = blockIdx.x * 16;
  const int n0 = wave * 16;
  const float* arow = agg + (size_t)(m0 + lm) * DIM;
  const float* brow = ldsW + (size_t)(n0 + lm) * DIM;  // B[k][n] = W[n][k]
  v8f acc = {};
#pragma unroll
  for (int k0 = 0; k0 < DIM; k0 += 32) {
    v16bf a = frag_a_f32(arow + k0, half);
    v16bf b = frag_b_f32(brow + k0, half);
    acc = WMMA_BF16(a, b, acc);
  }
  const int col = n0 + lm;
  const float bv = bias[col];
#pragma unroll
  for (int i = 0; i < 8; ++i) {
    const int row = m0 + i + half * 8;
    out[(size_t)row * DIM + col] = acc[i] + bv;
  }
}

// ---- eft[t] = concat(item_h[item2idx[t]], cat_h[cat2idx[t]]) ----
__global__ void eft_gather(const float* __restrict__ itemH, const float* __restrict__ catH,
                           const int* __restrict__ i2x, const int* __restrict__ c2x,
                           float* __restrict__ eft) {
  const int idx = blockIdx.x * blockDim.x + threadIdx.x;
  if (idx >= T_POS * D2) return;
  const int t = idx >> 8, d = idx & 255;
  const float v = (d < DIM) ? itemH[(size_t)i2x[t] * DIM + d]
                            : catH[(size_t)c2x[t] * DIM + (d - DIM)];
  eft[idx] = v;
}

__global__ void last_pos(const int* __restrict__ seg, int* __restrict__ last_idx) {
  const int t = blockIdx.x * blockDim.x + threadIdx.x;
  if (t < T_POS) atomicMax(&last_idx[seg[t]], t);
}

// ---- alpha[t] = sigmoid(eft@w1^T + b1 + eft[last]@w2^T) @ q^T + qb ----
__global__ void attn_alpha_wmma(const float* __restrict__ eft, const int* __restrict__ seg,
                                const int* __restrict__ last_idx,
                                const float* __restrict__ w1W, const float* __restrict__ w1b,
                                const float* __restrict__ w2W, const float* __restrict__ qW,
                                const float* __restrict__ qb, float* __restrict__ alpha) {
  const int lane = threadIdx.x & 31;
  const int wave = threadIdx.x >> 5;
  const int half = lane >> 4;
  const int lm   = lane & 15;
  const int t0 = blockIdx.x * 16;
  const int n0 = wave * 32;
  const int tr = t0 + lm;
  const float* xrow = eft + (size_t)tr * D2;
  const float* lrow = eft + (size_t)last_idx[seg[tr]] * D2;
  const float* b0w1 = w1W + (size_t)(n0 + lm) * D2;
  const float* b1w1 = w1W + (size_t)(n0 + 16 + lm) * D2;
  const float* b0w2 = w2W + (size_t)(n0 + lm) * D2;
  const float* b1w2 = w2W + (size_t)(n0 + 16 + lm) * D2;
  v8f acc0 = {}, acc1 = {};
#pragma unroll
  for (int k0 = 0; k0 < D2; k0 += 32) {
    v16bf a = frag_a_f32(xrow + k0, half);
    acc0 = WMMA_BF16(a, frag_b_f32(b0w1 + k0, half), acc0);
    acc1 = WMMA_BF16(a, frag_b_f32(b1w1 + k0, half), acc1);
    v16bf a2 = frag_a_f32(lrow + k0, half);
    acc0 = WMMA_BF16(a2, frag_b_f32(b0w2 + k0, half), acc0);
    acc1 = WMMA_BF16(a2, frag_b_f32(b1w2 + k0, half), acc1);
  }
  __shared__ float rowsum[16];
  if (threadIdx.x < 16) rowsum[threadIdx.x] = 0.0f;
  __syncthreads();
  const int c0 = n0 + lm, c1 = n0 + 16 + lm;
  const float bb0 = w1b[c0], bb1 = w1b[c1];
  const float q0 = qW[c0], q1 = qW[c1];
#pragma unroll
  for (int i = 0; i < 8; ++i) {
    const float s0 = 1.0f / (1.0f + __expf(-(acc0[i] + bb0)));
    const float s1 = 1.0f / (1.0f + __expf(-(acc1[i] + bb1)));
    float v = s0 * q0 + s1 * q1;
#pragma unroll
    for (int off = 1; off < 16; off <<= 1) v += __shfl_xor(v, off, 32);
    if (lm == 0) atomicAdd(&rowsum[i + half * 8], v);
  }
  __syncthreads();
  if (threadIdx.x < 16) alpha[t0 + threadIdx.x] = rowsum[threadIdx.x] + qb[0];
}

__device__ inline int lower_bound_i(const int* a, int n, int key) {
  int lo = 0, hi = n;
  while (lo < hi) { const int mid = (lo + hi) >> 1; if (a[mid] < key) lo = mid + 1; else hi = mid; }
  return lo;
}

__global__ void sess_sum(const float* __restrict__ eft, const float* __restrict__ alpha,
                         const int* __restrict__ seg, float* __restrict__ sg) {
  const int s = blockIdx.x;
  __shared__ int bounds[2];
  if (threadIdx.x == 0) bounds[0] = lower_bound_i(seg, T_POS, s);
  if (threadIdx.x == 1) bounds[1] = lower_bound_i(seg, T_POS, s + 1);
  __syncthreads();
  const int lo = bounds[0], hi = bounds[1];
  const int d = threadIdx.x;
  float acc = 0.0f;
  for (int t = lo; t < hi; ++t) acc += alpha[t] * eft[(size_t)t * D2 + d];
  sg[(size_t)s * D2 + d] = acc;
}

__global__ void sg_to_bf16(const float* __restrict__ sg, __bf16* __restrict__ sgb) {
  const int i = blockIdx.x * blockDim.x + threadIdx.x;
  if (i < S_SESS * D2) sgb[i] = (__bf16)sg[i];
}

__global__ void build_allb(const float* __restrict__ itemT, const float* __restrict__ catT,
                           const int* __restrict__ c4i, __bf16* __restrict__ allb) {
  const int idx = blockIdx.x * blockDim.x + threadIdx.x;
  if (idx >= V_ALL * D2) return;
  const int v = idx >> 8, d = idx & 255;
  const float f = (d < DIM) ? itemT[(size_t)v * DIM + d]
                            : catT[(size_t)c4i[v] * DIM + (d - DIM)];
  allb[idx] = (__bf16)f;
}

// ---- out[2048,100000] = sg_bf16 @ all_bf16^T  (block: 32x256, wave: 16x64) ----
// A tile (32 rows x 256 K bf16 = 16KB, contiguous) staged into LDS by the TDM
// (tensor_load_to_lds + TENSORcnt); B stays global (51MB bf16 -> L2-resident).
__global__ void final_gemm_wmma(const __bf16* __restrict__ A, const __bf16* __restrict__ B,
                                float* __restrict__ C) {
  __shared__ __bf16 ldsA[32 * D2];  // 16 KB
  const int lane = threadIdx.x & 31;
  const int wave = threadIdx.x >> 5;

  if (wave == 0) {
    const unsigned lbase = lds_off(&ldsA[0]);
    const unsigned long long gaddr =
        (unsigned long long)(size_t)A + (unsigned long long)blockIdx.y * (32u * D2 * 2u);
    // D# group 0: count=1 | lds_addr | global_addr[56:0] | type=2
    v4u g0;
    g0.x = 1u;                                   // count = 1, is_restore=0, gather off
    g0.y = lbase;                                // lds_addr (bytes)
    g0.z = (unsigned)(gaddr & 0xFFFFFFFFu);      // global_addr[31:0]
    g0.w = ((unsigned)(gaddr >> 32) & 0x01FFFFFFu) | (2u << 30);  // addr[56:32] | type=2
    // D# group 1: data_size=8B; tensor_dim0 = tile_dim0 = 2048 (x8B = 16KB); 1 row
    v8u g1;
    g1.s0 = 0x00030000u;  // workgroup_mask=0, data_size=3 (8B)
    g1.s1 = 0x08000000u;  // tensor_dim0[15:0] = 0x800 in bits [63:48]
    g1.s2 = 0x00010000u;  // tensor_dim0[31:16]=0 ; tensor_dim1[15:0]=1
    g1.s3 = 0x08000000u;  // tensor_dim1[31:16]=0 ; tile_dim0 = 0x800
    g1.s4 = 0x00000001u;  // tile_dim1 = 1 ; tile_dim2 = 0
    g1.s5 = 0x00000800u;  // tensor_dim0_stride[31:0] = 2048
    g1.s6 = 0x08000000u;  // stride0[47:32]=0 ; tensor_dim1_stride[15:0] = 0x800
    g1.s7 = 0u;
    asm volatile("tensor_load_to_lds %0, %1" :: "s"(g0), "s"(g1) : "memory");
    __builtin_amdgcn_s_wait_tensorcnt(0);
  }
  __syncthreads();

  const int half = lane >> 4;
  const int lm   = lane & 15;
  const int mloc = (wave >> 2) * 16;             // 0 or 16 within the staged 32 rows
  const int m0   = blockIdx.y * 32 + mloc;
  const int n0   = blockIdx.x * 256 + (wave & 3) * 64;
  const __bf16* arow  = ldsA + (size_t)(mloc + lm) * D2;
  const __bf16* brow0 = B + (size_t)min(n0 + lm,      V_ALL - 1) * D2;
  const __bf16* brow1 = B + (size_t)min(n0 + 16 + lm, V_ALL - 1) * D2;
  const __bf16* brow2 = B + (size_t)min(n0 + 32 + lm, V_ALL - 1) * D2;
  const __bf16* brow3 = B + (size_t)min(n0 + 48 + lm, V_ALL - 1) * D2;
  v8f acc0 = {}, acc1 = {}, acc2 = {}, acc3 = {};
#pragma unroll
  for (int k0 = 0; k0 < D2; k0 += 32) {
    v16bf a = frag_a_bf16(arow + k0, half);
    acc0 = WMMA_BF16(a, frag_b_bf16(brow0 + k0, half), acc0);
    acc1 = WMMA_BF16(a, frag_b_bf16(brow1 + k0, half), acc1);
    acc2 = WMMA_BF16(a, frag_b_bf16(brow2 + k0, half), acc2);
    acc3 = WMMA_BF16(a, frag_b_bf16(brow3 + k0, half), acc3);
  }
#pragma unroll
  for (int i = 0; i < 8; ++i) {
    const int row = m0 + i + half * 8;
    float* crow = C + (size_t)row * V_ALL;
    int c;
    c = n0 + lm;      if (c < V_ALL) crow[c] = acc0[i];
    c = n0 + 16 + lm; if (c < V_ALL) crow[c] = acc1[i];
    c = n0 + 32 + lm; if (c < V_ALL) crow[c] = acc2[i];
    c = n0 + 48 + lm; if (c < V_ALL) crow[c] = acc3[i];
  }
}

extern "C" void kernel_launch(void* const* d_in, const int* in_sizes, int n_in,
                              void* d_out, int out_size, void* d_ws, size_t ws_size,
                              hipStream_t stream) {
  (void)in_sizes; (void)n_in; (void)out_size; (void)ws_size;
  const int*   items     = (const int*)d_in[0];
  const int*   cats      = (const int*)d_in[1];
  const int*   item2idx  = (const int*)d_in[2];
  const int*   cat2idx   = (const int*)d_in[3];
  const int*   item_e    = (const int*)d_in[4];   // [2, E_I]
  const int*   cat_e     = (const int*)d_in[5];   // [2, E_C]
  const float* item_e_w  = (const float*)d_in[6];
  const float* cat_e_w   = (const float*)d_in[7];
  const int*   seg       = (const int*)d_in[8];
  const int*   cat4item  = (const int*)d_in[9];
  const float* item_tab  = (const float*)d_in[10];
  const float* cat_tab   = (const float*)d_in[11];
  const float* iW        = (const float*)d_in[12];
  const float* ib        = (const float*)d_in[13];
  const float* cW        = (const float*)d_in[14];
  const float* cb        = (const float*)d_in[15];
  const float* w1W       = (const float*)d_in[16];
  const float* w1b       = (const float*)d_in[17];
  const float* w2W       = (const float*)d_in[18];
  const float* qW        = (const float*)d_in[19];
  const float* qb        = (const float*)d_in[20];

  char* ws = (char*)d_ws;
  float*  item_agg = (float*)(ws + 0UL);          // 33.5 MB
  float*  cat_agg  = (float*)(ws + 33554432UL);   //  4.2 MB
  float*  item_h   = (float*)(ws + 37748736UL);   // 33.5 MB
  float*  cat_h    = (float*)(ws + 71303168UL);   //  4.2 MB
  float*  eft      = (float*)(ws + 75497472UL);   // 104.9 MB
  float*  alpha    = (float*)(ws + 180355072UL);  //  0.4 MB
  int*    last_idx = (int*)  (ws + 180764672UL);  //  8 KB
  float*  sg       = (float*)(ws + 180772864UL);  //  2.1 MB
  __bf16* sgb      = (__bf16*)(ws + 182870016UL); //  1.0 MB
  __bf16* allb     = (__bf16*)(ws + 183918592UL); // 51.2 MB
  float*  out      = (float*)d_out;

  hipMemsetAsync(item_agg, 0, (size_t)N_I * DIM * 4, stream);
  hipMemsetAsync(cat_agg,  0, (size_t)N_C * DIM * 4, stream);
  hipMemsetAsync(last_idx, 0, (size_t)S_SESS * 4, stream);

  gnn_scatter<<<E_I / 8, 256, 0, stream>>>(item_tab, items, item_e, item_e + E_I,
                                           item_e_w, item_agg, E_I);
  gnn_scatter<<<E_C / 8, 256, 0, stream>>>(cat_tab, cats, cat_e, cat_e + E_C,
                                           cat_e_w, cat_agg, E_C);
  gnn_linear_wmma<<<N_I / 16, 256, 0, stream>>>(item_agg, iW, ib, item_h);
  gnn_linear_wmma<<<N_C / 16, 256, 0, stream>>>(cat_agg, cW, cb, cat_h);
  eft_gather<<<(T_POS * D2) / 256, 256, 0, stream>>>(item_h, cat_h, item2idx, cat2idx, eft);
  last_pos<<<T_POS / 256, 256, 0, stream>>>(seg, last_idx);
  attn_alpha_wmma<<<T_POS / 16, 256, 0, stream>>>(eft, seg, last_idx,
                                                  w1W, w1b, w2W, qW, qb, alpha);
  sess_sum<<<S_SESS, 256, 0, stream>>>(eft, alpha, seg, sg);
  sg_to_bf16<<<(S_SESS * D2) / 256, 256, 0, stream>>>(sg, sgb);
  build_allb<<<(V_ALL * D2) / 256, 256, 0, stream>>>(item_tab, cat_tab, cat4item, allb);
  final_gemm_wmma<<<dim3((V_ALL + 255) / 256, S_SESS / 32), 256, 0, stream>>>(sgb, allb, out);
}
